// DeepFactorizationMachineModel_PolarSim_Align_24026047054762
// MI455X (gfx1250) — matrix operand order repaired and checked by reading, not compiled, not run on hardware
//
#include <hip/hip_runtime.h>
#include <hip/hip_bf16.h>
#include <math.h>

typedef __attribute__((ext_vector_type(16))) __bf16 v16bf;
typedef __attribute__((ext_vector_type(8)))  float  v8f;
typedef __attribute__((ext_vector_type(4)))  unsigned int u32x4;
typedef __attribute__((ext_vector_type(8)))  int i32x8;
typedef __attribute__((ext_vector_type(4)))  int i32x4;

#define FDS   39
#define EMB   64
#define BATCH 16384
#define VPF   26000
#define HDIM  400
#define KIN   (FDS*EMB)   // 2496
#define BN_INV 0.9999950000374997f

struct B32x8 { uint4 lo, hi; };

static __device__ __forceinline__ v16bf make_frag(uint4 lo, uint4 hi) {
  B32x8 u{lo, hi};
  return __builtin_bit_cast(v16bf, u);
}

static __device__ __forceinline__ unsigned pack2(__bf16 a, __bf16 b) {
  unsigned short ua = __builtin_bit_cast(unsigned short, a);
  unsigned short ub = __builtin_bit_cast(unsigned short, b);
  return (unsigned)ua | ((unsigned)ub << 16);
}

// Issue a TDM 2-D tile load: 64 rows x 32 bf16, row stride K elements, into LDS.
// tensor_dim0 = validK clamps the K tail (hardware zero-fills OOB reads).
static __device__ __forceinline__ void tdm_load_a_tile(const __bf16* gsrc,
                                                       unsigned ldsOff,
                                                       int K, int validK) {
  unsigned long long ga = (unsigned long long)(uintptr_t)gsrc;
  u32x4 g0;
  g0[0] = 1u;                                              // count=1, user D#
  g0[1] = ldsOff;                                          // lds_addr (bytes)
  g0[2] = (unsigned)(ga & 0xffffffffu);                    // global_addr[31:0]
  g0[3] = (unsigned)((ga >> 32) & 0x01ffffffu) | (2u << 30); // addr[56:32]|type=2
  i32x8 g1;
  g1[0] = (1 << 16);                                       // data_size=2B, mask=0
  g1[1] = (int)((validK & 0xffff) << 16);                  // tensor_dim0[15:0]
  g1[2] = (int)(((unsigned)validK >> 16) & 0xffffu) | (64 << 16); // dim0[31:16] | tensor_dim1=64
  g1[3] = (32 << 16);                                      // tensor_dim1[31:16]=0 | tile_dim0=32
  g1[4] = 64;                                              // tile_dim1=64, tile_dim2=0
  g1[5] = K;                                               // tensor_dim0_stride[31:0]
  g1[6] = 0;                                               // stride[47:32], dim1_stride lo
  g1[7] = 0;
  i32x4 z4; z4[0] = 0; z4[1] = 0; z4[2] = 0; z4[3] = 0;
  i32x8 z8; for (int i = 0; i < 8; ++i) z8[i] = 0;
  __builtin_amdgcn_tensor_load_to_lds(g0, g1, z4, z4, z8, 0);
}

// ---------------------------------------------------------------------------
// Kernel 1: embedding gather + sparse gate + per-field 64x64 WMMA transform
//           + FM 2nd-order term + 1st-order linear term.
// One workgroup (4 waves) handles a 64-batch tile; each wave owns 16 batches.
// ---------------------------------------------------------------------------
__launch_bounds__(128)
__global__ void k_embed_fm(const int* __restrict__ x,
                           const float* __restrict__ emb,
                           const float* __restrict__ lin_table,
                           const float* __restrict__ lin_bias,
                           const float* __restrict__ sparse_var,
                           const float* __restrict__ Wt,
                           const float* __restrict__ bt,
                           __bf16* __restrict__ exg,
                           float* __restrict__ fmlin)
{
  __shared__ __align__(32) float  svf[EMB];
  __shared__ __align__(32) __bf16 wt[EMB][EMB];       // Wt[f][o][i] in bf16
  __shared__ __align__(32) __bf16 ex[4][16][EMB];     // per-wave A tiles
  __shared__ float red[4][8][32];

  const int tid      = threadIdx.x;
  const int wave     = tid >> 5;
  const int lane     = tid & 31;
  const int l16      = lane & 15;
  const int halfSel  = lane >> 4;          // 0 or 1
  const int tileBase = blockIdx.x * 64;

  v8f S[4], Q[4];
  for (int nf = 0; nf < 4; ++nf)
    for (int i = 0; i < 8; ++i) { S[nf][i] = 0.f; Q[nf][i] = 0.f; }

  for (int f = 0; f < FDS; ++f) {
    __syncthreads();   // protect wt/svf reuse across iterations
    // stage sparse gate for this field
    if (tid < EMB) {
      float s = sparse_var[f*EMB + tid];
      float v = 1.f / (1.f + __expf(-15.f * s));
      svf[tid] = (v > 0.001f) ? v : 0.f;
    }
    // stage Wt[f] (row o, col i) -> bf16 LDS
    {
      int o  = tid >> 1;
      int i0 = (tid & 1) * 32;
      const float4* src = (const float4*)(Wt + (size_t)f*EMB*EMB + (size_t)o*EMB + i0);
      for (int j = 0; j < 8; ++j) {
        float4 v = src[j];
        uint2 pp;
        pp.x = pack2((__bf16)v.x, (__bf16)v.y);
        pp.y = pack2((__bf16)v.z, (__bf16)v.w);
        *(uint2*)&wt[o][i0 + j*4] = pp;
      }
    }
    __syncthreads();
    // gather 16 rows per wave, gate, convert -> LDS + global embed_x (bf16)
    {
      int r  = lane >> 1;
      int c0 = (lane & 1) * 32;
      int b  = tileBase + wave*16 + r;
      int col = x[b*FDS + f];
      size_t row = (size_t)col + (size_t)f * VPF;
      const float4* src = (const float4*)(emb + row*EMB + c0);
      __bf16* gdst = exg + (size_t)b*KIN + f*EMB + c0;
      for (int j = 0; j < 8; ++j) {
        float4 v = src[j];
        int c = c0 + j*4;
        v.x *= svf[c+0]; v.y *= svf[c+1]; v.z *= svf[c+2]; v.w *= svf[c+3];
        uint2 pp;
        pp.x = pack2((__bf16)v.x, (__bf16)v.y);
        pp.y = pack2((__bf16)v.z, (__bf16)v.w);
        *(uint2*)&ex[wave][r][c] = pp;
        *(uint2*)(gdst + j*4)    = pp;
      }
    }
    __syncthreads();
    // trans(16x64) = ex(16x64) x Wt[f]^T, C init = bt[f]
    v8f T[4];
    for (int nf = 0; nf < 4; ++nf) {
      float cb = bt[f*EMB + nf*16 + l16];
      for (int i = 0; i < 8; ++i) T[nf][i] = cb;
    }
    for (int ks = 0; ks < 2; ++ks) {
      int k0 = ks * 32;
      const __bf16* arow = &ex[wave][l16][0];
      int kh = k0 + halfSel * 16;
      uint4 alo = *(const uint4*)(arow + kh);
      uint4 ahi = *(const uint4*)(arow + kh + 8);
      v16bf a = make_frag(alo, ahi);
      for (int nf = 0; nf < 4; ++nf) {
        v16bf bm = *(const v16bf*)&wt[nf*16 + l16][k0 + halfSel*16];
        T[nf] = __builtin_amdgcn_wmma_f32_16x16x32_bf16(
                    false, a, false, bm, (short)0, T[nf], false, false);
      }
    }
    for (int nf = 0; nf < 4; ++nf) {
      S[nf] += T[nf];
      Q[nf] += T[nf] * T[nf];
    }
  }

  // FM: fm[b] = 0.5 * sum_o (S^2 - Q).  C layout: M = r + 8*halfSel, N = l16.
  for (int r = 0; r < 8; ++r) {
    float p = 0.f;
    for (int nf = 0; nf < 4; ++nf) { float s = S[nf][r]; p += s*s - Q[nf][r]; }
    red[wave][r][lane] = p;
  }
  __syncthreads();
  if (lane < 16) {
    float sum = 0.f;
    if (lane < 8) { for (int j = 0;  j < 16; ++j) sum += red[wave][lane][j]; }
    else          { for (int j = 16; j < 32; ++j) sum += red[wave][lane-8][j]; }
    float fm = 0.5f * sum;
    int b = tileBase + wave*16 + lane;
    float lt = lin_bias[0];
    for (int f = 0; f < FDS; ++f) {
      int col = x[b*FDS + f];
      lt += lin_table[(size_t)col + (size_t)f*VPF];
    }
    fmlin[b] = lt + fm;
  }
}

// ---------------------------------------------------------------------------
// Kernels 2-4: bf16 WMMA GEMM, Out = relu((A@W + bias) * BN_INV*g + be)
// BM=64 BN=64 BK=32, 4 waves/workgroup, each wave 16x64 of output.
// A tiles arrive via the Tensor Data Mover (TENSOR_LOAD_TO_LDS, TENSORcnt),
// double-buffered against the WMMA pipeline; W tiles are transposed +
// converted fp32->bf16 on the fly by the VALU path.
// ---------------------------------------------------------------------------
__launch_bounds__(128)
__global__ void k_gemm_bn_relu(const __bf16* __restrict__ A,
                               const float* __restrict__ W,
                               const float* __restrict__ bias,
                               const float* __restrict__ gamma,
                               const float* __restrict__ beta,
                               __bf16* __restrict__ Out,
                               int M, int K, int N)
{
  __shared__ __align__(32) __bf16 la[2][64][32];   // [buf][m][k]  (TDM dest)
  __shared__ __align__(32) __bf16 lb[2][64][32];   // [buf][n][k]  (transposed)

  const int tid     = threadIdx.x;
  const int wave    = tid >> 5;
  const int lane    = tid & 31;
  const int l16     = lane & 15;
  const int halfSel = lane >> 4;
  const int m0 = blockIdx.x * 64;
  const int n0 = blockIdx.y * 64;

  v8f acc[4];
  for (int nf = 0; nf < 4; ++nf)
    for (int i = 0; i < 8; ++i) acc[nf][i] = 0.f;

  const int ksteps = (K + 31) / 32;

  // stage B tile for step kt into buffer buf (zero-padded at K/N edges)
  auto stageB = [&](int kt, int buf) {
    int k0 = kt * 32;
    int n  = tid & 63;
    int kb = (tid >> 6) * 16;
    for (int j = 0; j < 16; ++j) {
      int k = k0 + kb + j;
      float v = (k < K && (n0 + n) < N) ? W[(size_t)k*N + n0 + n] : 0.f;
      lb[buf][n][kb + j] = (__bf16)v;
    }
    if ((kt + 1) < ksteps)  // prefetch next W tile (L2 resident)
      __builtin_prefetch(W + (size_t)(k0 + 32)*N + n0 + n, 0, 0);
  };

  // prologue: TDM A(0), stage B(0)
  if (tid < 32) {
    tdm_load_a_tile(A + (size_t)m0 * K,
                    (unsigned)(uintptr_t)(void*)&la[0][0][0], K,
                    (K < 32 ? K : 32));
  }
  stageB(0, 0);
  if (tid < 32) __builtin_amdgcn_s_wait_tensorcnt(0);
  __syncthreads();

  for (int kt = 0; kt < ksteps; ++kt) {
    int buf = kt & 1;
    // kick off next tile's staging while this tile computes
    if (kt + 1 < ksteps) {
      int k0n = (kt + 1) * 32;
      if (tid < 32) {
        int vk = K - k0n; if (vk > 32) vk = 32;
        tdm_load_a_tile(A + (size_t)m0 * K + k0n,
                        (unsigned)(uintptr_t)(void*)&la[buf ^ 1][0][0], K, vk);
      }
      stageB(kt + 1, buf ^ 1);
    }
    // WMMA on current tile
    {
      const __bf16* arow = &la[buf][wave*16 + l16][halfSel*16];
      uint4 alo = *(const uint4*)(arow);
      uint4 ahi = *(const uint4*)(arow + 8);
      v16bf a = make_frag(alo, ahi);
      for (int nf = 0; nf < 4; ++nf) {
        v16bf bm = *(const v16bf*)&lb[buf][nf*16 + l16][halfSel*16];
        acc[nf] = __builtin_amdgcn_wmma_f32_16x16x32_bf16(
                      false, a, false, bm, (short)0, acc[nf], false, false);
      }
    }
    // drain TDM for the tile we just prefetched, then publish to all waves
    if (tid < 32) __builtin_amdgcn_s_wait_tensorcnt(0);
    __syncthreads();
  }

  // fused bias + eval-BN + ReLU epilogue, bf16 out
  for (int nf = 0; nf < 4; ++nf) {
    int n = n0 + nf*16 + l16;
    if (n < N) {
      float sc = BN_INV * gamma[n];
      float bb = bias[n];
      float sh = beta[n];
      for (int r = 0; r < 8; ++r) {
        int m = m0 + wave*16 + r + 8*halfSel;
        float y = (acc[nf][r] + bb) * sc + sh;
        y = y > 0.f ? y : 0.f;
        Out[(size_t)m*N + n] = (__bf16)y;
      }
    }
  }
}

// ---------------------------------------------------------------------------
// Kernel 5: out[b] = sigmoid(fmlin[b] + h3[b,:] . Wout + bout)
// ---------------------------------------------------------------------------
__launch_bounds__(256)
__global__ void k_final(const __bf16* __restrict__ h3,
                        const float* __restrict__ Wout,
                        const float* __restrict__ bout,
                        const float* __restrict__ fmlin,
                        float* __restrict__ out)
{
  int b = blockIdx.x * blockDim.x + threadIdx.x;
  if (b >= BATCH) return;
  const __bf16* row = h3 + (size_t)b * HDIM;
  float acc = 0.f;
  for (int n = 0; n < HDIM; ++n) acc += (float)row[n] * Wout[n];
  float z = fmlin[b] + acc + bout[0];
  out[b] = 1.f / (1.f + __expf(-z));
}

// ---------------------------------------------------------------------------
extern "C" void kernel_launch(void* const* d_in, const int* in_sizes, int n_in,
                              void* d_out, int out_size, void* d_ws, size_t ws_size,
                              hipStream_t stream)
{
  const int*   x    = (const int*)  d_in[0];
  const float* emb  = (const float*)d_in[1];
  const float* lint = (const float*)d_in[2];
  const float* linb = (const float*)d_in[3];
  const float* sv   = (const float*)d_in[4];
  const float* Wt   = (const float*)d_in[5];
  const float* bt   = (const float*)d_in[6];
  const float* W1   = (const float*)d_in[7];
  const float* b1   = (const float*)d_in[8];
  const float* g1w  = (const float*)d_in[9];
  const float* be1  = (const float*)d_in[10];
  const float* W2   = (const float*)d_in[11];
  const float* b2   = (const float*)d_in[12];
  const float* g2w  = (const float*)d_in[13];
  const float* be2  = (const float*)d_in[14];
  const float* W3   = (const float*)d_in[15];
  const float* b3   = (const float*)d_in[16];
  const float* g3w  = (const float*)d_in[17];
  const float* be3  = (const float*)d_in[18];
  const float* Wout = (const float*)d_in[19];
  const float* bout = (const float*)d_in[20];
  float* out = (float*)d_out;

  char* ws = (char*)d_ws;
  size_t off = 0;
  auto wsAlloc = [&](size_t bytes) -> void* {
    void* p = ws + off;
    off = (off + bytes + 255) & ~(size_t)255;
    return p;
  };
  __bf16* exg   = (__bf16*)wsAlloc((size_t)BATCH * KIN  * sizeof(__bf16)); // 82 MB
  float*  fmlin = (float*) wsAlloc((size_t)BATCH * sizeof(float));
  __bf16* h1    = (__bf16*)wsAlloc((size_t)BATCH * HDIM * sizeof(__bf16));
  __bf16* h2    = (__bf16*)wsAlloc((size_t)BATCH * HDIM * sizeof(__bf16));
  __bf16* h3    = (__bf16*)wsAlloc((size_t)BATCH * HDIM * sizeof(__bf16));

  k_embed_fm<<<BATCH/64, 128, 0, stream>>>(x, emb, lint, linb, sv, Wt, bt, exg, fmlin);

  dim3 gdim(BATCH/64, (HDIM + 63) / 64);
  k_gemm_bn_relu<<<gdim, 128, 0, stream>>>(exg, W1, b1, g1w, be1, h1, BATCH, KIN,  HDIM);
  k_gemm_bn_relu<<<gdim, 128, 0, stream>>>(h1,  W2, b2, g2w, be2, h2, BATCH, HDIM, HDIM);
  k_gemm_bn_relu<<<gdim, 128, 0, stream>>>(h2,  W3, b3, g3w, be3, h3, BATCH, HDIM, HDIM);

  k_final<<<BATCH/256, 256, 0, stream>>>(h3, Wout, bout, fmlin, out);
}
